// NpuQuantizationLinear_78374563217503
// MI455X (gfx1250) — 8-bit matrix operand order repaired and checked by reading, compiled, not run on hardware
//
#include <hip/hip_runtime.h>
#include <stdint.h>

#define M_DIM 8192
#define K_DIM 4096
#define N_DIM 4096
#define KB64  (K_DIM / 64)

#define BM 128
#define BN 128
#define BK 128
#define LDA 136

typedef int   v8i __attribute__((ext_vector_type(8)));
typedef int   v4i __attribute__((ext_vector_type(4)));
typedef int   v2i __attribute__((ext_vector_type(2)));
typedef float v4f __attribute__((ext_vector_type(4)));

typedef __attribute__((address_space(1))) v4i g_v4i;
typedef __attribute__((address_space(3))) v4i l_v4i;

__device__ __forceinline__ void async_cp_b128(const uint8_t* g, uint8_t* l) {
#if __has_builtin(__builtin_amdgcn_global_load_async_to_lds_b128)
    __builtin_amdgcn_global_load_async_to_lds_b128(
        (g_v4i*)(uintptr_t)g, (l_v4i*)(uintptr_t)l, 0, 0);
#else
    const uint32_t lo = (uint32_t)(uintptr_t)l;
    asm volatile("global_load_async_to_lds_b128 %0, %1, off"
                 :: "v"(lo), "v"(g) : "memory");
#endif
}

__device__ __forceinline__ void wait_async0() {
#if __has_builtin(__builtin_amdgcn_s_wait_asynccnt)
    __builtin_amdgcn_s_wait_asynccnt(0);
#else
    asm volatile("s_wait_asynccnt 0x0" ::: "memory");
#endif
}

__global__ void quantize_x_kernel(const float* __restrict__ x,
                                  const float* __restrict__ act_scale,
                                  const float* __restrict__ act_offset,
                                  uint32_t* __restrict__ qx) {
    const float s = act_scale[0];
    const float o = act_offset[0];
    const size_t i = (size_t)blockIdx.x * blockDim.x + threadIdx.x;
    const v4f xv = ((const v4f*)x)[i];
    uint32_t packed = 0;
#pragma unroll
    for (int j = 0; j < 4; ++j) {
        float v = rintf(xv[j] * s + o);
        v = fminf(fmaxf(v, -128.0f), 127.0f);
        int q = (int)v;
        packed |= ((uint32_t)(q & 0xff)) << (8 * j);
    }
    qx[i] = packed;
}

__global__ void pack_w_kernel(const int8_t* __restrict__ w,
                              uint32_t* __restrict__ wp) {
    const uint32_t d    = blockIdx.x * blockDim.x + threadIdx.x;
    const uint32_t frag = d >> 8;
    const uint32_t off  = (d & 255u) << 2;
    const uint32_t half = off >> 9;
    const uint32_t lofs = off & 511u;
    const uint32_t l    = lofs >> 4;
    const uint32_t b    = lofs & 15u;
    const uint32_t koff = ((l >> 4) << 4) + (half << 5) + b;
    const uint32_t nb   = frag / KB64;
    const uint32_t kb   = frag % KB64;
    const uint32_t gcol = nb * 16u + (l & 15u);
    const size_t base = (size_t)(kb * 64u + koff) * N_DIM + gcol;
    const uint32_t b0 = (uint8_t)w[base];
    const uint32_t b1 = (uint8_t)w[base + (size_t)N_DIM];
    const uint32_t b2 = (uint8_t)w[base + (size_t)2 * N_DIM];
    const uint32_t b3 = (uint8_t)w[base + (size_t)3 * N_DIM];
    wp[d] = b0 | (b1 << 8) | (b2 << 16) | (b3 << 24);
}

__global__ __launch_bounds__(256)
void gemm_i8_kernel(const uint8_t* __restrict__ qx,
                    const uint8_t* __restrict__ wp,
                    const int*     __restrict__ bias,
                    const float*   __restrict__ scale,
                    float*         __restrict__ out)
{
    __shared__ uint8_t lds_a[2][BM * LDA];
    __shared__ uint8_t lds_b[2][16 * 1024];

    const int t    = threadIdx.x;
    const int lane = t & 31;
    const int wv   = t >> 5;
    const int wm   = wv >> 2;
    const int wn   = wv & 3;
    const int half = lane >> 4;
    const int lr   = lane & 15;

    const int bn = blockIdx.x;
    const int bm = blockIdx.y;

    v8i acc[4][2];
#pragma unroll
    for (int i = 0; i < 4; ++i)
#pragma unroll
        for (int j = 0; j < 2; ++j)
#pragma unroll
            for (int v = 0; v < 8; ++v) acc[i][j][v] = 0;

    const int arow = t >> 1;
    const int acol = (t & 1) * 64;
    const size_t a_gbase = (size_t)(bm * BM + arow) * K_DIM + acol;
    const int a_loff = arow * LDA + acol;

    const int bfrag  = t >> 4;
    const int binner = t & 15;
    const int fnb = bfrag >> 1;
    const int fkb = bfrag & 1;
    const size_t b_gfrag0 = (size_t)((bn * 8 + fnb) * KB64 + fkb);
    const int b_loff = (bfrag << 10) + (binner << 6);

    {
        const uint8_t* ag = qx + a_gbase;
        const uint8_t* bg = wp + (b_gfrag0 << 10) + (binner << 6);
#pragma unroll
        for (int i = 0; i < 4; ++i) {
            async_cp_b128(ag + 16 * i, &lds_a[0][a_loff + 16 * i]);
            async_cp_b128(bg + 16 * i, &lds_b[0][b_loff + 16 * i]);
        }
    }

    for (int k0 = 0; k0 < K_DIM; k0 += BK) {
        const int p = (k0 >> 7) & 1;

        wait_async0();
        __syncthreads();

        if (k0 + BK < K_DIM) {
            const uint8_t* ag = qx + a_gbase + k0 + BK;
            const uint8_t* bg = wp + ((b_gfrag0 + (size_t)((k0 + BK) >> 6)) << 10)
                                   + (binner << 6);
#pragma unroll
            for (int i = 0; i < 4; ++i) {
                async_cp_b128(ag + 16 * i, &lds_a[1 - p][a_loff + 16 * i]);
                async_cp_b128(bg + 16 * i, &lds_b[1 - p][b_loff + 16 * i]);
            }
        }

#pragma unroll
        for (int ks = 0; ks < BK; ks += 64) {
            v8i bfr[2];
#pragma unroll
            for (int tn = 0; tn < 2; ++tn) {
                const int f = ((wn * 2 + tn) << 1) + (ks >> 6);
                const uint8_t* pb = &lds_b[p][(f << 10) + (lane << 4)];
                v4i lo = *(const v4i*)(pb);
                v4i hi = *(const v4i*)(pb + 512);
#pragma unroll
                for (int v = 0; v < 4; ++v) { bfr[tn][v] = lo[v]; bfr[tn][v + 4] = hi[v]; }
            }
            v8i afr[4];
#pragma unroll
            for (int tm = 0; tm < 4; ++tm) {
                const uint8_t* pa = &lds_a[p][(wm * 64 + tm * 16 + lr) * LDA + ks + half * 16];
                v2i d0 = *(const v2i*)(pa +  0);
                v2i d1 = *(const v2i*)(pa +  8);
                v2i d2 = *(const v2i*)(pa + 32);
                v2i d3 = *(const v2i*)(pa + 40);
                afr[tm][0] = d0[0]; afr[tm][1] = d0[1];
                afr[tm][2] = d1[0]; afr[tm][3] = d1[1];
                afr[tm][4] = d2[0]; afr[tm][5] = d2[1];
                afr[tm][6] = d3[0]; afr[tm][7] = d3[1];
            }
#pragma unroll
            for (int tm = 0; tm < 4; ++tm)
#pragma unroll
                for (int tn = 0; tn < 2; ++tn)
                    acc[tm][tn] = __builtin_amdgcn_wmma_i32_16x16x64_iu8(
                         true, afr[tm],
                         true, bfr[tn],
                        acc[tm][tn],
                         false,  false);
        }
    }

#pragma unroll
    for (int tn = 0; tn < 2; ++tn) {
        const int gcol = bn * BN + wn * 32 + tn * 16 + lr;
        const int   bv = bias[gcol];
        const float sv = scale[gcol];
#pragma unroll
        for (int tm = 0; tm < 4; ++tm) {
            const int grow0 = bm * BM + wm * 64 + tm * 16 + half * 8;
            float* op = out + (size_t)grow0 * N_DIM + gcol;
#pragma unroll
            for (int v = 0; v < 8; ++v)
                op[(size_t)v * N_DIM] = (float)(acc[tm][tn][v] + bv) * sv;
        }
    }
}

extern "C" void kernel_launch(void* const* d_in, const int* in_sizes, int n_in,
                              void* d_out, int out_size, void* d_ws, size_t ws_size,
                              hipStream_t stream) {
    const float*  x          = (const float*)d_in[0];
    const int8_t* w          = (const int8_t*)d_in[1];
    const float*  act_scale  = (const float*)d_in[2];
    const float*  act_offset = (const float*)d_in[3];
    const float*  deq_scale  = (const float*)d_in[4];
    const int*    bias_i32   = (const int*)d_in[5];
    float*        out        = (float*)d_out;

    uint8_t* qx = (uint8_t*)d_ws;
    uint8_t* wp = qx + (size_t)M_DIM * K_DIM;
    (void)in_sizes; (void)n_in; (void)out_size; (void)ws_size;

    quantize_x_kernel<<<(M_DIM * K_DIM / 4) / 256, 256, 0, stream>>>(
        x, act_scale, act_offset, (uint32_t*)qx);

    pack_w_kernel<<<(K_DIM * N_DIM / 4) / 256, 256, 0, stream>>>(
        w, (uint32_t*)wp);

    dim3 grid(N_DIM / BN, M_DIM / BM);
    gemm_i8_kernel<<<grid, 256, 0, stream>>>(qx, wp, bias_i32, deq_scale, out);
}
